// MixLlamaMLP_27616639713537
// MI455X (gfx1250) — matrix operand order repaired and checked by reading, not compiled, not run on hardware
//
#include <hip/hip_runtime.h>
#include <hip/hip_bf16.h>
#include <stdint.h>

// ---------------------------------------------------------------------------
// LLaMA MLP: y = (silu(x@Wg) * (x@Wu)) @ Wd
// M = B*S = 8192, H = 4096, I = 11008, fp32 in/out.
// Compute-bound -> bf16 WMMA (v_wmma_f32_16x16x32_bf16, f32 accumulate).
// Kernel 1: gate+up fused across wave groups, SiLU*up combine via LDS.
// Kernel 2: down proj; bf16 activation tile staged by the Tensor Data Mover
// (tensor_load_to_lds + s_wait_tensorcnt) straight into padded LDS.
// Double-buffered LDS, register-staged weight loads, one barrier per K-step,
// GROUP_M grid swizzle for L2 weight reuse.
// ---------------------------------------------------------------------------

typedef __attribute__((ext_vector_type(16))) __bf16 v16bf;
typedef __attribute__((ext_vector_type(8)))  __bf16 v8bf;
typedef __attribute__((ext_vector_type(8)))  float  v8f;
typedef __attribute__((ext_vector_type(4)))  int    i32x4;
typedef __attribute__((ext_vector_type(8)))  int    i32x8;

namespace {

constexpr int HD = 4096;    // hidden dim
constexpr int ID = 11008;   // intermediate dim
constexpr int MT = 8192;    // tokens = B*S

constexpr int BM   = 128;   // block tile M
constexpr int BN1  = 64;    // block tile N (kernel 1, per weight matrix)
constexpr int BN2  = 128;   // block tile N (kernel 2)
constexpr int BK   = 32;    // block tile K == WMMA K
constexpr int LDK  = BK + 8;          // padded LDS row (80B)
constexpr int ATILE  = BM  * LDK;     // LDS A tile elems
constexpr int BTILE1 = BN1 * LDK;     // LDS B tile elems (kernel 1)
constexpr int BTILE2 = BN2 * LDK;     // LDS B tile elems (kernel 2)
constexpr int GROUP_M = 8;            // grid swizzle group

__device__ __forceinline__ unsigned int rne_bf16_hi(unsigned int u) {
  return u + 0x7FFFu + ((u >> 16) & 1u);   // bf16 RNE, result in [31:16]
}

__device__ __forceinline__ unsigned short f2bf(float f) {
  return (unsigned short)(rne_bf16_hi(__float_as_uint(f)) >> 16);
}

// Pack two floats into packed bf16 (lo -> [15:0], hi -> [31:16]).
__device__ __forceinline__ unsigned int pk_bf16(float lo, float hi) {
#if __has_builtin(__builtin_amdgcn_cvt_pk_bf16_f32)
  typedef __attribute__((ext_vector_type(2))) __bf16 v2bf;
  union { v2bf v; unsigned int u; } t;
  t.v = __builtin_amdgcn_cvt_pk_bf16_f32(lo, hi);
  return t.u;
#else
  return __builtin_amdgcn_perm(rne_bf16_hi(__float_as_uint(hi)),
                               rne_bf16_hi(__float_as_uint(lo)),
                               0x07060302u);
#endif
}

__device__ __forceinline__ v16bf frag_cat(v8bf lo, v8bf hi) {
  union { v16bf v; v8bf h[2]; } t;
  t.h[0] = lo; t.h[1] = hi;
  return t.v;
}

// A fragment (16x32 bf16 MxK): lane%16 = row; lanes 0-15 hold K 0..7,16..23,
// lanes 16-31 hold K 8..15,24..31. LDS layout: [m][k] padded.
__device__ __forceinline__ v16bf load_frag_a(const unsigned short* s, int row, int lane) {
  const int c = (lane < 16) ? 0 : 8;
  const v8bf lo = *reinterpret_cast<const v8bf*>(s + row * LDK + c);
  const v8bf hi = *reinterpret_cast<const v8bf*>(s + row * LDK + c + 16);
  return frag_cat(lo, hi);
}

// B fragment (32x16 bf16 KxN): lane%16 = col; lanes 0-15 hold K 0..7,16..23,
// lanes 16-31 hold K 8..15,24..31. LDS layout: transposed [n][k] padded.
__device__ __forceinline__ v16bf load_frag_b(const unsigned short* s, int col, int lane) {
  const int c = (lane < 16) ? 0 : 8;
  const v8bf lo = *reinterpret_cast<const v8bf*>(s + col * LDK + c);
  const v8bf hi = *reinterpret_cast<const v8bf*>(s + col * LDK + c + 16);
  return frag_cat(lo, hi);
}

__device__ __forceinline__ v8f wmma_bf16(v16bf a, v16bf b, v8f c) {
  return __builtin_amdgcn_wmma_f32_16x16x32_bf16(
      false, a, false, b, (short)0, c, false, false);
}

// Generic pointer -> LDS byte offset (ISA 10.2: low 32 address bits).
__device__ __forceinline__ unsigned lds_off(const void* p) {
  return (unsigned)(uintptr_t)(const __attribute__((address_space(3))) void*)p;
}

// Tensor Data Mover: async-load a 2D tile of 2-byte elements into LDS with
// hardware row padding (pad_interval = 16 DWORDs, pad_amount = 4 DWORDs ->
// 64B row + 16B pad == the LDK=40 padded layout). D# per ISA 8.3/8.4.
// One wave issues it; completion tracked by TENSORcnt.
__device__ __forceinline__ void tdm_load_tile_bf16(
    const void* gtile, unsigned lds_byte, unsigned row_stride_elems) {
  const unsigned long long ga = (unsigned long long)(uintptr_t)gtile;
  i32x4 g0;
  g0[0] = 1;                                         // count=1 (valid), user
  g0[1] = (int)lds_byte;                             // lds_addr
  g0[2] = (int)(unsigned)(ga & 0xFFFFFFFFu);         // global_addr[31:0]
  g0[3] = (int)(((ga >> 32) & 0x01FFFFFFu) | (2u << 30)); // addr[56:32]|type=2
  i32x8 g1;
  g1[0] = (1 << 16)                                  // data_size = 2 bytes
        | (1 << 20)                                  // pad_enable
        | (3 << 22)                                  // pad_interval: 16 DW
        | (3 << 25);                                 // pad_amount: 4 DW
  g1[1] = (int)((unsigned)ID << 16);                 // tensor_dim0[15:0]
  g1[2] = (int)(((unsigned)ID >> 16) | ((unsigned)MT << 16));  // d0 hi | d1 lo
  g1[3] = (int)(((unsigned)MT >> 16) | ((unsigned)BK << 16));  // d1 hi | tile_dim0
  g1[4] = (int)BM;                                   // tile_dim1 (tile_dim2=0)
  g1[5] = (int)row_stride_elems;                     // tensor_dim0_stride lo
  g1[6] = 0;                                         // stride hi | dim1_stride
  g1[7] = 0;
  asm volatile("tensor_load_to_lds %0, %1" :: "s"(g0), "s"(g1) : "memory");
}

// CUTLASS-style swizzle: GROUP_M M-blocks per N sweep -> activation slab stays
// L2-resident while weights stream once per group.
__device__ __forceinline__ void tile_coords(int gridN, int& bm, int& bn) {
  const int bid = blockIdx.y * gridDim.x + blockIdx.x;
  const int per_group = GROUP_M * gridN;
  const int g = bid / per_group;
  const int r = bid % per_group;
  bm = g * GROUP_M + (r % GROUP_M);
  bn = r / GROUP_M;
}

// ---- register staging ------------------------------------------------------
__device__ __forceinline__ void load_a_regs(float4 (&r)[4], const float* __restrict__ x,
                                            int m0, int k0, int ldx, int t) {
#pragma unroll
  for (int i = 0; i < 4; ++i) {
    const int idx = t + i * 256;
    const int m  = idx >> 3;
    const int kv = (idx & 7) * 4;
    r[i] = *reinterpret_cast<const float4*>(x + (size_t)(m0 + m) * ldx + k0 + kv);
  }
}

__device__ __forceinline__ void store_a_lds(unsigned short* As, const float4 (&r)[4], int t) {
#pragma unroll
  for (int i = 0; i < 4; ++i) {
    const int idx = t + i * 256;
    const int m  = idx >> 3;
    const int kv = (idx & 7) * 4;
    uint2 p;
    p.x = pk_bf16(r[i].x, r[i].y);
    p.y = pk_bf16(r[i].z, r[i].w);
    *reinterpret_cast<uint2*>(As + m * LDK + kv) = p;
  }
}

// Weight tile BKxW f32 row-major [k][n]: 4x4 block per thread.
template <int W>
__device__ __forceinline__ void load_w_regs(float4 (&r)[4], const float* __restrict__ w,
                                            int k0, int n0, int ldw, int tt) {
  constexpr int NG = W / 4;
  const int kl = (tt / NG) * 4;
  const int nl = (tt % NG) * 4;
#pragma unroll
  for (int i = 0; i < 4; ++i)
    r[i] = *reinterpret_cast<const float4*>(w + (size_t)(k0 + kl + i) * ldw + n0 + nl);
}

template <int W>
__device__ __forceinline__ void store_w_lds(unsigned short* Bs, const float4 (&r)[4], int tt) {
  constexpr int NG = W / 4;
  const int kl = (tt / NG) * 4;
  const int nl = (tt % NG) * 4;
  const float c0[4] = {r[0].x, r[0].y, r[0].z, r[0].w};
  const float c1[4] = {r[1].x, r[1].y, r[1].z, r[1].w};
  const float c2[4] = {r[2].x, r[2].y, r[2].z, r[2].w};
  const float c3[4] = {r[3].x, r[3].y, r[3].z, r[3].w};
#pragma unroll
  for (int j = 0; j < 4; ++j) {
    uint2 p;
    p.x = pk_bf16(c0[j], c1[j]);
    p.y = pk_bf16(c2[j], c3[j]);
    *reinterpret_cast<uint2*>(Bs + (nl + j) * LDK + kl) = p;
  }
}

} // namespace

// ---------------------------------------------------------------------------
// Kernel 1: act = silu(x@Wg) * (x@Wu), bf16 out (d_ws).
// Block tile 128x64; waves 0-3 gate, waves 4-7 up; wave tile 64x32.
// ---------------------------------------------------------------------------
__global__ __launch_bounds__(256) void mlp_gateup_kernel(
    const float* __restrict__ x, const float* __restrict__ wg,
    const float* __restrict__ wu, unsigned short* __restrict__ act) {
  __shared__ __align__(16) unsigned char smem[2 * ATILE * 2 + 4 * BTILE1 * 2];
  unsigned short* AsB = reinterpret_cast<unsigned short*>(smem);
  unsigned short* GsB = AsB + 2 * ATILE;
  unsigned short* UsB = GsB + 2 * BTILE1;
  float* Xs = reinterpret_cast<float*>(smem);   // epilogue exchange (128x65)

  const int t    = threadIdx.x;
  const int lane = t & 31;
  const int wave = t >> 5;
  const int wg4  = wave & 3;
  const int wm   = wg4 & 1;         // 64 rows
  const int wn   = wg4 >> 1;        // 32 cols
  const bool is_up = wave >= 4;

  int bm, bn;
  tile_coords(ID / BN1, bm, bn);
  const int m0 = bm * BM;
  const int n0 = bn * BN1;

  unsigned short* MyB = is_up ? UsB : GsB;          // compute source
  unsigned short* StB = (t < 128) ? GsB : UsB;      // staging target
  const float* stw    = (t < 128) ? wg : wu;
  const int tt = t & 127;

  v8f acc[4][2];
#pragma unroll
  for (int i = 0; i < 4; ++i)
#pragma unroll
    for (int j = 0; j < 2; ++j) acc[i][j] = (v8f)0.0f;

  float4 ar[4], br[4];

  load_a_regs(ar, x, m0, 0, HD, t);
  load_w_regs<BN1>(br, stw, 0, n0, ID, tt);
  store_a_lds(AsB, ar, t);
  store_w_lds<BN1>(StB, br, tt);
  __syncthreads();

  int cur = 0;
  for (int k0 = 0; k0 < HD; k0 += BK) {
    const bool has_next = (k0 + BK) < HD;
    if (has_next) {
      load_a_regs(ar, x, m0, k0 + BK, HD, t);
      load_w_regs<BN1>(br, stw, k0 + BK, n0, ID, tt);
    }

    const unsigned short* As = AsB + cur * ATILE;
    const unsigned short* Bs = MyB + cur * BTILE1;

    v16bf af[4];
#pragma unroll
    for (int i = 0; i < 4; ++i)
      af[i] = load_frag_a(As, wm * 64 + i * 16 + (lane & 15), lane);
#pragma unroll
    for (int j = 0; j < 2; ++j) {
      const v16bf bf = load_frag_b(Bs, wn * 32 + j * 16 + (lane & 15), lane);
#pragma unroll
      for (int i = 0; i < 4; ++i) acc[i][j] = wmma_bf16(af[i], bf, acc[i][j]);
    }

    if (has_next) {
      store_a_lds(AsB + (cur ^ 1) * ATILE, ar, t);
      store_w_lds<BN1>(StB + (cur ^ 1) * BTILE1, br, tt);
    }
    __syncthreads();
    cur ^= 1;
  }

  // ---- epilogue: combine gate & up across wave groups via LDS -------------
  const int cn    = lane & 15;
  const int rbase = (lane < 16) ? 0 : 8;
  constexpr int XLD = BN1 + 1;

  if (is_up) {
#pragma unroll
    for (int i = 0; i < 4; ++i)
#pragma unroll
      for (int j = 0; j < 2; ++j) {
        const int col = wn * 32 + j * 16 + cn;
#pragma unroll
        for (int r = 0; r < 8; ++r) {
          const int row = wm * 64 + i * 16 + rbase + r;
          Xs[row * XLD + col] = acc[i][j][r];
        }
      }
  }
  __syncthreads();
  if (!is_up) {
#pragma unroll
    for (int i = 0; i < 4; ++i)
#pragma unroll
      for (int j = 0; j < 2; ++j) {
        const int col = wn * 32 + j * 16 + cn;
#pragma unroll
        for (int r = 0; r < 8; ++r) {
          const int row = wm * 64 + i * 16 + rbase + r;
          const float g = acc[i][j][r];
          const float u = Xs[row * XLD + col];
          const float s = g / (1.0f + __expf(-g));   // SiLU
          act[(size_t)(m0 + row) * ID + (n0 + col)] = f2bf(s * u);
        }
      }
  }
}

// ---------------------------------------------------------------------------
// Kernel 2: y = act @ Wd  (act bf16 [MT][ID], Wd f32 [ID][HD], y f32 d_out).
// Block tile 128x128, 8 waves 2(M) x 4(N). A tile staged by the TDM
// (tensor_load_to_lds, wave 0) with hardware row padding; weights staged
// through registers with on-the-fly f32->bf16 conversion.
// ---------------------------------------------------------------------------
__global__ __launch_bounds__(256) void mlp_down_kernel(
    const unsigned short* __restrict__ act, const float* __restrict__ wd,
    float* __restrict__ y) {
  __shared__ __align__(16) unsigned char smem[(2 * ATILE + 2 * BTILE2) * 2];
  unsigned short* AsB = reinterpret_cast<unsigned short*>(smem);
  unsigned short* BsB = AsB + 2 * ATILE;

  const int t    = threadIdx.x;
  const int lane = t & 31;
  const int wave = t >> 5;
  const int wm   = wave & 1;   // 64 rows
  const int wn   = wave >> 1;  // 32 cols
  const bool tdm_wave = (wave == 0);

  int bm, bn;
  tile_coords(HD / BN2, bm, bn);
  const int m0 = bm * BM;
  const int n0 = bn * BN2;

  const unsigned lds_a0 = lds_off(AsB);
  const unsigned lds_a1 = lds_off(AsB + ATILE);
  const unsigned short* atile = act + (size_t)m0 * ID;

  v8f acc[4][2];
#pragma unroll
  for (int i = 0; i < 4; ++i)
#pragma unroll
    for (int j = 0; j < 2; ++j) acc[i][j] = (v8f)0.0f;

  float4 br[4];

  if (tdm_wave) tdm_load_tile_bf16(atile, lds_a0, ID);
  load_w_regs<BN2>(br, wd, 0, n0, HD, t);
  store_w_lds<BN2>(BsB, br, t);
  if (tdm_wave) __builtin_amdgcn_s_wait_tensorcnt(0);
  __syncthreads();

  int cur = 0;
  for (int k0 = 0; k0 < ID; k0 += BK) {
    const bool has_next = (k0 + BK) < ID;
    if (has_next) {
      if (tdm_wave)
        tdm_load_tile_bf16(atile + (k0 + BK), cur ? lds_a0 : lds_a1, ID);
      load_w_regs<BN2>(br, wd, k0 + BK, n0, HD, t);
    }

    const unsigned short* As = AsB + cur * ATILE;
    const unsigned short* Bs = BsB + cur * BTILE2;

    v16bf af[4];
#pragma unroll
    for (int i = 0; i < 4; ++i)
      af[i] = load_frag_a(As, wm * 64 + i * 16 + (lane & 15), lane);
#pragma unroll
    for (int j = 0; j < 2; ++j) {
      const v16bf bf = load_frag_b(Bs, wn * 32 + j * 16 + (lane & 15), lane);
#pragma unroll
      for (int i = 0; i < 4; ++i) acc[i][j] = wmma_bf16(af[i], bf, acc[i][j]);
    }

    if (has_next) store_w_lds<BN2>(BsB + (cur ^ 1) * BTILE2, br, t);
    if (tdm_wave) __builtin_amdgcn_s_wait_tensorcnt(0);
    __syncthreads();
    cur ^= 1;
  }

  const int cn    = lane & 15;
  const int rbase = (lane < 16) ? 0 : 8;
#pragma unroll
  for (int i = 0; i < 4; ++i) {
#pragma unroll
    for (int j = 0; j < 2; ++j) {
      const int gcol = n0 + wn * 32 + j * 16 + cn;
#pragma unroll
      for (int r = 0; r < 8; ++r) {
        const int grow = m0 + wm * 64 + i * 16 + rbase + r;
        y[(size_t)grow * HD + gcol] = acc[i][j][r];
      }
    }
  }
}

// ---------------------------------------------------------------------------
extern "C" void kernel_launch(void* const* d_in, const int* in_sizes, int n_in,
                              void* d_out, int out_size, void* d_ws, size_t ws_size,
                              hipStream_t stream) {
  const float* x  = (const float*)d_in[0];  // [4,2048,4096]
  const float* wg = (const float*)d_in[1];  // [4096,11008]
  const float* wu = (const float*)d_in[2];  // [4096,11008]
  const float* wd = (const float*)d_in[3];  // [11008,4096]
  float* y = (float*)d_out;                 // [4,2048,4096]

  unsigned short* act = (unsigned short*)d_ws;  // bf16 [8192,11008] (~180 MB)

  dim3 block(256, 1, 1);
  dim3 grid1(ID / BN1, MT / BM, 1);  // (172, 64), swizzled in-kernel
  mlp_gateup_kernel<<<grid1, block, 0, stream>>>(x, wg, wu, act);

  dim3 grid2(HD / BN2, MT / BM, 1);  // (32, 64), swizzled in-kernel
  mlp_down_kernel<<<grid2, block, 0, stream>>>(act, wd, y);
}
